// GNNModelDGL_85177791414880
// MI455X (gfx1250) — matrix operand order repaired and checked by reading, not compiled, not run on hardware
//
#include <hip/hip_runtime.h>
#include <hip/hip_bf16.h>

// ---------------------------------------------------------------------------
// GNN (2-layer GAT + mean pool + ArcFace) for gfx1250.
// Dense GEMMs: v_wmma_f32_16x16x32_bf16, one wave per 16x64 stripe
// (A fragment loaded once per k-step, reused across 4 B tiles).
// Edge softmax: 3-pass atomics (seg-max / seg-sum / weighted aggregate).
// ---------------------------------------------------------------------------

typedef __attribute__((ext_vector_type(16))) __bf16 v16bf;
typedef __attribute__((ext_vector_type(8)))  float  v8f;

union Frag { uint4 q[2]; v16bf v; };

__device__ __forceinline__ unsigned short f2bf(float f) {
    unsigned u = __float_as_uint(f);
    unsigned r = u + 0x7FFFu + ((u >> 16) & 1u);   // round-to-nearest-even
    return (unsigned short)(r >> 16);
}

// monotone encoding so unsigned atomicMax == float max
__device__ __forceinline__ unsigned fenc(float f) {
    unsigned u = __float_as_uint(f);
    return (u & 0x80000000u) ? ~u : (u | 0x80000000u);
}
__device__ __forceinline__ float fdec(unsigned k) {
    unsigned u = (k & 0x80000000u) ? (k & 0x7FFFFFFFu) : ~k;
    return __uint_as_float(u);
}

// ------------------------------ prep ---------------------------------------

__global__ void cvt_bf16(const float* __restrict__ in, unsigned short* __restrict__ out, long n) {
    long i = (long)blockIdx.x * blockDim.x + threadIdx.x;
    if (i < n) out[i] = f2bf(in[i]);
}

// W[K][Nn] (row-major) -> Wt[Nn][K] bf16 (column-major B for WMMA)
__global__ void transpose_w_bf16(const float* __restrict__ W, unsigned short* __restrict__ Wt,
                                 int K, int Nn) {
    long i = (long)blockIdx.x * blockDim.x + threadIdx.x;
    if (i >= (long)K * Nn) return;
    int k = (int)(i / Nn), n = (int)(i - (long)k * Nn);
    Wt[(size_t)n * K + k] = f2bf(W[i]);
}

// ------------------------------ WMMA GEMM ----------------------------------
// C[M][Nn] = A[M][K] (bf16 row-major) x Bt[Nn][K] (bf16 col-major-of-B)
// One wave per 16x64 C stripe: A fragment loaded once per k-step,
// reused across 4 B fragments -> 4 WMMAs per k-step.
__global__ void wmma_gemm_bf16(const unsigned short* __restrict__ A,
                               const unsigned short* __restrict__ Bt,
                               float* __restrict__ C, int M, int Nn, int K) {
    const int lane = threadIdx.x & 31;
    const int wave = blockIdx.x * (blockDim.x >> 5) + (threadIdx.x >> 5);
    const int nstripes = Nn >> 6;                   // 64-wide stripes
    const int mt = wave / nstripes;
    const int nt = wave - mt * nstripes;
    if (mt * 16 >= M) return;

    const int hsel = lane >> 4;        // 0: lanes 0-15, 1: lanes 16-31
    const int l15  = lane & 15;
    const unsigned short* arow = A + (size_t)(mt * 16 + l15) * K;
    const unsigned short* brow[4];
#pragma unroll
    for (int j = 0; j < 4; ++j)
        brow[j] = Bt + (size_t)(nt * 64 + j * 16 + l15) * K;

    v8f acc0 = {}, acc1 = {}, acc2 = {}, acc3 = {};
    for (int k0 = 0; k0 < K; k0 += 32) {
        Frag fa;
        // A 16x32 bf16, contiguous k order chosen here (same as B): lanes 0-15 hold K=0..15, lanes 16-31 hold K=16..31
        const uint4* ap = (const uint4*)(arow + k0 + hsel * 16);
        fa.q[0] = ap[0];
        fa.q[1] = ap[1];                            // +8 elements
        Frag fb0, fb1, fb2, fb3;
        // B 32x16 bf16, same contiguous k order as A: lanes 0-15 hold K=0..15, lanes 16-31 hold K=16..31
        const uint4* bp0 = (const uint4*)(brow[0] + k0 + hsel * 16);
        const uint4* bp1 = (const uint4*)(brow[1] + k0 + hsel * 16);
        const uint4* bp2 = (const uint4*)(brow[2] + k0 + hsel * 16);
        const uint4* bp3 = (const uint4*)(brow[3] + k0 + hsel * 16);
        fb0.q[0] = bp0[0]; fb0.q[1] = bp0[1];
        fb1.q[0] = bp1[0]; fb1.q[1] = bp1[1];
        fb2.q[0] = bp2[0]; fb2.q[1] = bp2[1];
        fb3.q[0] = bp3[0]; fb3.q[1] = bp3[1];
        acc0 = __builtin_amdgcn_wmma_f32_16x16x32_bf16(false, fa.v, false, fb0.v,
                                                       (short)0, acc0, false, false);
        acc1 = __builtin_amdgcn_wmma_f32_16x16x32_bf16(false, fa.v, false, fb1.v,
                                                       (short)0, acc1, false, false);
        acc2 = __builtin_amdgcn_wmma_f32_16x16x32_bf16(false, fa.v, false, fb2.v,
                                                       (short)0, acc2, false, false);
        acc3 = __builtin_amdgcn_wmma_f32_16x16x32_bf16(false, fa.v, false, fb3.v,
                                                       (short)0, acc3, false, false);
    }
    // C layout: VGPR r -> row 8*hsel + r, col = l15
    float* cbase = C + (size_t)(mt * 16 + hsel * 8) * Nn + nt * 64 + l15;
#pragma unroll
    for (int r = 0; r < 8; ++r) {
        float* cr = cbase + (size_t)r * Nn;
        cr[0]  = acc0[r];
        cr[16] = acc1[r];
        cr[32] = acc2[r];
        cr[48] = acc3[r];
    }
}

// ------------------------------ attention scores ---------------------------

// layer1: el/er [N,8], feat1 [N, 8*32]
__global__ void attn_scores1(const float* __restrict__ feat, const float* __restrict__ al,
                             const float* __restrict__ ar, float* __restrict__ el,
                             float* __restrict__ er, int nNodes) {
    int i = blockIdx.x * blockDim.x + threadIdx.x;   // n*8 + h
    if (i >= nNodes * 8) return;
    int n = i >> 3, h = i & 7;
    const float* f = feat + (size_t)n * 256 + h * 32;
    const float* a = al + h * 32;
    const float* b = ar + h * 32;
    float sl = 0.f, sr = 0.f;
#pragma unroll 8
    for (int d = 0; d < 32; ++d) { float v = f[d]; sl += v * a[d]; sr += v * b[d]; }
    el[i] = sl; er[i] = sr;
}

// layer2: el/er [N], feat2 [N,256]
__global__ void attn_scores2(const float* __restrict__ feat, const float* __restrict__ al,
                             const float* __restrict__ ar, float* __restrict__ el,
                             float* __restrict__ er, int nNodes) {
    int n = blockIdx.x * blockDim.x + threadIdx.x;
    if (n >= nNodes) return;
    const float* f = feat + (size_t)n * 256;
    float sl = 0.f, sr = 0.f;
    for (int d = 0; d < 256; ++d) { float v = f[d]; sl += v * al[d]; sr += v * ar[d]; }
    el[n] = sl; er[n] = sr;
}

// ------------------------------ edge softmax (3 passes) --------------------

__global__ void edge_max(const int* __restrict__ src, const int* __restrict__ dst,
                         const float* __restrict__ el, const float* __restrict__ er,
                         unsigned* __restrict__ emax, int E, int H) {
    int e = blockIdx.x * blockDim.x + threadIdx.x;
    if (e >= E) return;
    int s = src[e], d = dst[e];
    for (int h = 0; h < H; ++h) {
        float v = el[s * H + h] + er[d * H + h];
        v = v > 0.f ? v : 0.2f * v;                 // leaky_relu(0.2)
        atomicMax(&emax[d * H + h], fenc(v));
    }
}

__global__ void edge_exp(const int* __restrict__ src, const int* __restrict__ dst,
                         const float* __restrict__ el, const float* __restrict__ er,
                         const unsigned* __restrict__ emax, float* __restrict__ alpha,
                         float* __restrict__ denom, int E, int H) {
    int e = blockIdx.x * blockDim.x + threadIdx.x;
    if (e >= E) return;
    int s = src[e], d = dst[e];
    for (int h = 0; h < H; ++h) {
        float v = el[s * H + h] + er[d * H + h];
        v = v > 0.f ? v : 0.2f * v;
        float ex = __expf(v - fdec(emax[d * H + h]));
        alpha[(size_t)e * H + h] = ex;
        atomicAdd(&denom[d * H + h], ex);
    }
}

// one wave per edge; 256 channels, 8 per lane
__global__ void edge_aggregate(const int* __restrict__ src, const int* __restrict__ dst,
                               const float* __restrict__ feat, const float* __restrict__ alpha,
                               const float* __restrict__ denom, float* __restrict__ agg,
                               int E, int H) {
    int e = blockIdx.x * (blockDim.x >> 5) + (threadIdx.x >> 5);
    if (e >= E) return;
    int lane = threadIdx.x & 31;
    int s = src[e], d = dst[e];
#pragma unroll
    for (int i = 0; i < 8; ++i) {
        int c = lane + 32 * i;
        int h = (c * H) >> 8;                       // H=8 -> c/32, H=1 -> 0
        float w = alpha[(size_t)e * H + h] / denom[(size_t)d * H + h];
        atomicAdd(&agg[(size_t)d * 256 + c], feat[(size_t)s * 256 + c] * w);
    }
}

// ------------------------------ pointwise ----------------------------------

__global__ void bias_elu(const float* __restrict__ agg, const float* __restrict__ b,
                         float* __restrict__ hout, unsigned short* __restrict__ hbf, long total) {
    long i = (long)blockIdx.x * blockDim.x + threadIdx.x;
    if (i >= total) return;
    int c = (int)(i & 255);
    float v = agg[i] + b[c];
    float r = v > 0.f ? v : (__expf(v) - 1.f);      // ELU
    hout[i] = r;
    hbf[i] = f2bf(r);
}

__global__ void pool_nodes(const float* __restrict__ agg2, const float* __restrict__ h,
                           const float* __restrict__ b2, const int* __restrict__ gid,
                           float* __restrict__ gsum, float* __restrict__ cnt, long total) {
    long i = (long)blockIdx.x * blockDim.x + threadIdx.x;
    if (i >= total) return;
    int n = (int)(i >> 8), c = (int)(i & 255);
    float v = agg2[i] + h[i] + b2[c];               // residual + bias (1 head -> mean is identity)
    int g = gid[n];
    atomicAdd(&gsum[(size_t)g * 256 + c], v);
    if (c == 0) atomicAdd(&cnt[g], 1.f);
}

__global__ void norm_rows_bf16(const float* __restrict__ x, const float* __restrict__ cnt,
                               unsigned short* __restrict__ out) {
    __shared__ float red[256];
    int g = blockIdx.x, c = threadIdx.x;
    float v = x[(size_t)g * 256 + c];
    if (cnt) v /= fmaxf(cnt[g], 1.f);
    red[c] = v * v;
    __syncthreads();
    for (int s = 128; s > 0; s >>= 1) {
        if (c < s) red[c] += red[c + s];
        __syncthreads();
    }
    float rn = rsqrtf(fmaxf(red[0], 1e-30f));
    out[(size_t)g * 256 + c] = f2bf(v * rn);
}

// ------------------------------ ArcFace head (WMMA) ------------------------
// cosine = ni[G,256] @ nw[NCpad,256]^T ; out = cos(acos(c) + m*1{label}) * S
// One wave per 16x64 stripe (A fragment reused across 4 B tiles).
__global__ void arcface_wmma(const unsigned short* __restrict__ ni,
                             const unsigned short* __restrict__ nw,
                             const int* __restrict__ labels, float* __restrict__ out,
                             int Gn, int NC, int NCpad) {
    const int lane = threadIdx.x & 31;
    const int wave = blockIdx.x * (blockDim.x >> 5) + (threadIdx.x >> 5);
    const int nstripes = NCpad >> 6;
    const int mt = wave / nstripes;
    const int nt = wave - mt * nstripes;
    if (mt * 16 >= Gn) return;
    const int hsel = lane >> 4;
    const int l15  = lane & 15;
    const unsigned short* arow = ni + (size_t)(mt * 16 + l15) * 256;
    const unsigned short* brow[4];
#pragma unroll
    for (int j = 0; j < 4; ++j)
        brow[j] = nw + (size_t)(nt * 64 + j * 16 + l15) * 256;

    v8f acc[4] = {{}, {}, {}, {}};
    for (int k0 = 0; k0 < 256; k0 += 32) {
        Frag fa;
        const uint4* ap = (const uint4*)(arow + k0 + hsel * 16);
        fa.q[0] = ap[0];
        fa.q[1] = ap[1];
#pragma unroll
        for (int j = 0; j < 4; ++j) {
            Frag fb;
            const uint4* bp = (const uint4*)(brow[j] + k0 + hsel * 16);
            fb.q[0] = bp[0];
            fb.q[1] = bp[1];
            acc[j] = __builtin_amdgcn_wmma_f32_16x16x32_bf16(
                false, fa.v, false, fb.v, (short)0, acc[j], false, false);
        }
    }
    int rowbase = mt * 16 + hsel * 8;
#pragma unroll
    for (int j = 0; j < 4; ++j) {
        int col = nt * 64 + j * 16 + l15;
        if (col < NC) {
#pragma unroll
            for (int r = 0; r < 8; ++r) {
                int g = rowbase + r;
                float c = fminf(1.f, fmaxf(-1.f, acc[j][r]));
                float m = (labels[g] == col) ? 0.5f : 0.f;
                out[(size_t)g * NC + col] = cosf(acosf(c) + m) * 4.0f;
            }
        }
    }
}

// ---------------------------------------------------------------------------

extern "C" void kernel_launch(void* const* d_in, const int* in_sizes, int n_in,
                              void* d_out, int out_size, void* d_ws, size_t ws_size,
                              hipStream_t stream) {
    const float* features = (const float*)d_in[0];
    const int*   src      = (const int*)d_in[1];
    const int*   dst      = (const int*)d_in[2];
    const int*   gids     = (const int*)d_in[3];
    const int*   labels   = (const int*)d_in[4];
    const float* W1  = (const float*)d_in[5];
    const float* al1 = (const float*)d_in[6];
    const float* ar1 = (const float*)d_in[7];
    const float* b1  = (const float*)d_in[8];
    const float* W2  = (const float*)d_in[9];
    const float* al2 = (const float*)d_in[10];
    const float* ar2 = (const float*)d_in[11];
    const float* b2  = (const float*)d_in[12];
    const float* arc_w = (const float*)d_in[13];

    const int IN  = 128, HID = 256;
    const int Nn  = in_sizes[0] / IN;           // 40000 nodes
    const int E   = in_sizes[1];                // 640000 edges
    const int Gn  = in_sizes[4];                // 128 graphs
    const int NC  = in_sizes[13] / HID;         // 1000 classes
    const int NCpad = (NC + 63) & ~63;          // 1024 (64-wide WMMA stripes)

    // ---- workspace carve-out (256B aligned) ----
    char* base = (char*)d_ws;
    size_t off = 0;
    auto alloc = [&](size_t bytes) -> void* {
        off = (off + 255) & ~(size_t)255;
        void* p = base + off;
        off += bytes;
        return p;
    };
    unsigned short* fbf16 = (unsigned short*)alloc((size_t)Nn * IN * 2);
    unsigned short* w1t   = (unsigned short*)alloc((size_t)HID * IN * 2);
    unsigned short* w2t   = (unsigned short*)alloc((size_t)HID * HID * 2);
    float* feat1 = (float*)alloc((size_t)Nn * HID * 4);   // reused as agg2 later
    float* agg1  = (float*)alloc((size_t)Nn * HID * 4);   // reused as feat2 later
    float* hbuf  = (float*)alloc((size_t)Nn * HID * 4);
    unsigned short* hbf = (unsigned short*)alloc((size_t)Nn * HID * 2);
    float* el1 = (float*)alloc((size_t)Nn * 8 * 4);
    float* er1 = (float*)alloc((size_t)Nn * 8 * 4);
    unsigned* emax1 = (unsigned*)alloc((size_t)Nn * 8 * 4);
    float* denom1 = (float*)alloc((size_t)Nn * 8 * 4);
    float* alpha1 = (float*)alloc((size_t)E * 8 * 4);
    float* el2 = (float*)alloc((size_t)Nn * 4);
    float* er2 = (float*)alloc((size_t)Nn * 4);
    unsigned* emax2 = (unsigned*)alloc((size_t)Nn * 4);
    float* denom2 = (float*)alloc((size_t)Nn * 4);
    float* alpha2 = (float*)alloc((size_t)E * 4);
    float* gsum = (float*)alloc((size_t)Gn * HID * 4);
    float* cnt  = (float*)alloc((size_t)Gn * 4);
    unsigned short* nib = (unsigned short*)alloc((size_t)Gn * HID * 2);
    unsigned short* nwb = (unsigned short*)alloc((size_t)NCpad * HID * 2);
    float* feat2 = agg1;       // alias: agg1 dead after bias_elu
    float* agg2  = feat1;      // alias: feat1 dead after layer-1 aggregation

    const int T = 256;
    auto blks = [](long n, int t) { return (int)((n + t - 1) / t); };

    // ---- zero accumulators (every call; deterministic) ----
    hipMemsetAsync(emax1, 0, (size_t)Nn * 8 * 4, stream);
    hipMemsetAsync(denom1, 0, (size_t)Nn * 8 * 4, stream);
    hipMemsetAsync(agg1, 0, (size_t)Nn * HID * 4, stream);
    hipMemsetAsync(emax2, 0, (size_t)Nn * 4, stream);
    hipMemsetAsync(denom2, 0, (size_t)Nn * 4, stream);
    hipMemsetAsync(gsum, 0, (size_t)Gn * HID * 4, stream);
    hipMemsetAsync(cnt, 0, (size_t)Gn * 4, stream);
    hipMemsetAsync(nwb, 0, (size_t)NCpad * HID * 2, stream);

    // ---- prep: bf16 activations + transposed bf16 weights ----
    cvt_bf16<<<blks((long)Nn * IN, T), T, 0, stream>>>(features, fbf16, (long)Nn * IN);
    transpose_w_bf16<<<blks((long)IN * HID, T), T, 0, stream>>>(W1, w1t, IN, HID);
    transpose_w_bf16<<<blks((long)HID * HID, T), T, 0, stream>>>(W2, w2t, HID, HID);

    // ---- layer 1 ----
    int gemm_blocks = ((Nn / 16) * (HID / 64) + 7) / 8;
    wmma_gemm_bf16<<<gemm_blocks, T, 0, stream>>>(fbf16, w1t, feat1, Nn, HID, IN);
    attn_scores1<<<blks((long)Nn * 8, T), T, 0, stream>>>(feat1, al1, ar1, el1, er1, Nn);
    edge_max<<<blks(E, T), T, 0, stream>>>(src, dst, el1, er1, emax1, E, 8);
    edge_exp<<<blks(E, T), T, 0, stream>>>(src, dst, el1, er1, emax1, alpha1, denom1, E, 8);
    edge_aggregate<<<(E + 7) / 8, T, 0, stream>>>(src, dst, feat1, alpha1, denom1, agg1, E, 8);
    bias_elu<<<blks((long)Nn * HID, T), T, 0, stream>>>(agg1, b1, hbuf, hbf, (long)Nn * HID);

    // ---- layer 2 (feat2 aliases agg1; agg2 aliases feat1) ----
    wmma_gemm_bf16<<<gemm_blocks, T, 0, stream>>>(hbf, w2t, feat2, Nn, HID, HID);
    attn_scores2<<<blks(Nn, T), T, 0, stream>>>(feat2, al2, ar2, el2, er2, Nn);
    hipMemsetAsync(agg2, 0, (size_t)Nn * HID * 4, stream);   // feat1 is dead now
    edge_max<<<blks(E, T), T, 0, stream>>>(src, dst, el2, er2, emax2, E, 1);
    edge_exp<<<blks(E, T), T, 0, stream>>>(src, dst, el2, er2, emax2, alpha2, denom2, E, 1);
    edge_aggregate<<<(E + 7) / 8, T, 0, stream>>>(src, dst, feat2, alpha2, denom2, agg2, E, 1);

    // ---- residual + bias + per-graph mean pool ----
    pool_nodes<<<blks((long)Nn * HID, T), T, 0, stream>>>(agg2, hbuf, b2, gids, gsum, cnt,
                                                          (long)Nn * HID);

    // ---- normalize embeddings + weights ----
    norm_rows_bf16<<<Gn, HID, 0, stream>>>(gsum, cnt, nib);
    norm_rows_bf16<<<NC, HID, 0, stream>>>(arc_w, nullptr, nwb);

    // ---- ArcFace cosine GEMM + margin ----
    int arc_blocks = ((Gn / 16) * (NCpad / 64) + 7) / 8;
    arcface_wmma<<<arc_blocks, T, 0, stream>>>(nib, nwb, labels, (float*)d_out, Gn, NC, NCpad);
}
